// GKanConvLayer_45208825757999
// MI455X (gfx1250) — matrix operand order repaired and checked by reading, not compiled, not run on hardware
//
#include <hip/hip_runtime.h>
#include <hip/hip_bf16.h>
#include <math.h>

// ---------------------------------------------------------------------------
// KAN-GCN layer for MI455X (gfx1250, wave32, WMMA).
//
// Roofline: HBM traffic ~64MB (~3us at 23.3TB/s); everything fits in 192MB L2.
// Dominant cost is edge message passing (~820MB L2 gather/scatter + 102M f32
// atomics). Matmuls (16 GFLOP) lower to v_wmma_f32_16x16x32_bf16 and are
// effectively free. This version uses a 64-node M-tile per block (4 M-tiles
// per wave, 4 accumulators) so each weight (B) fragment fetched from L1/L2 is
// reused 4x, cutting weight-read traffic from ~0.9GB to ~225MB.
// ---------------------------------------------------------------------------

#define N_IN    64
#define N_OUT   64
#define N_COEF  8
#define KSPL    (N_IN * N_COEF)   // 512 spline K-dim
#define TILE_M  64                // nodes per block
#define M_TILES 4                 // 16-row WMMA tiles per block

typedef __attribute__((ext_vector_type(16))) __bf16 v16bf;
typedef __attribute__((ext_vector_type(8)))  __bf16 v8bf;
typedef __attribute__((ext_vector_type(8)))  float  v8f;

// ----------------------------- helpers -------------------------------------

__global__ void gk_zero_f32(float* __restrict__ p, int n) {
  int i = blockIdx.x * blockDim.x + threadIdx.x;
  int stride = gridDim.x * blockDim.x;
  for (; i < n; i += stride) p[i] = 0.0f;
}

__global__ void gk_cvt_f32_bf16(const float* __restrict__ s, __bf16* __restrict__ d, int n) {
  int i = blockIdx.x * blockDim.x + threadIdx.x;
  if (i < n) d[i] = (__bf16)s[i];
}

__global__ void gk_degree(const int* __restrict__ ei, float* __restrict__ deg, int nE) {
  int e = blockIdx.x * blockDim.x + threadIdx.x;
  if (e < nE) atomicAdd(&deg[ei[nE + e]], 1.0f);
}

__global__ void gk_dinv(const float* __restrict__ deg, float* __restrict__ dinv, int nN) {
  int i = blockIdx.x * blockDim.x + threadIdx.x;
  if (i < nN) {
    float d = deg[i];
    dinv[i] = (d > 0.0f) ? rsqrtf(d) : 0.0f;
  }
}

// One wave per edge: lane handles features 2*lane, 2*lane+1 (coalesced b64).
__global__ void gk_scatter(const float* __restrict__ x, const int* __restrict__ ei,
                           const float* __restrict__ dinv, float* __restrict__ agg, int nE) {
  int lane = threadIdx.x & 31;
  int e = blockIdx.x * (blockDim.x >> 5) + (threadIdx.x >> 5);
  if (e >= nE) return;
  int src = ei[e];
  int dst = ei[nE + e];
  float c = dinv[src] * dinv[dst];
  float2 v = ((const float2*)(x + (size_t)src * N_IN))[lane];
  float* ar = agg + (size_t)dst * N_IN + lane * 2;
  atomicAdd(ar,     v.x * c);
  atomicAdd(ar + 1, v.y * c);
}

// --------------------------- fragment loads --------------------------------
// 16-bit A (16x32) layout: lane L holds row M=L%16;
//   elems 0..7  = K[k0 + hi*8 .. +8),  elems 8..15 = K[k0 + 16 + hi*8 .. +8)
__device__ __forceinline__ v16bf gk_load_a(const __bf16* row, int k0, int hi) {
  v8bf lo = *(const v8bf*)(row + k0 + hi * 8);
  v8bf hv = *(const v8bf*)(row + k0 + 16 + hi * 8);
  return __builtin_shufflevector(lo, hv, 0,1,2,3,4,5,6,7,8,9,10,11,12,13,14,15);
}
// 16-bit B (32x16) layout: lane L holds col N=L%16;
//   elems 0..7  = K[k0 + hi*8 .. +8),  elems 8..15 = K[k0 + 16 + hi*8 .. +8)
__device__ __forceinline__ v16bf gk_load_b(const __bf16* row, int k0, int hi) {
  const __bf16* p = row + k0 + hi * 8;
  v8bf lo = *(const v8bf*)(p);
  v8bf hv = *(const v8bf*)(p + 16);
  return __builtin_shufflevector(lo, hv, 0,1,2,3,4,5,6,7,8,9,10,11,12,13,14,15);
}

// ------------------------------ fused KAN ----------------------------------
// Block = 64 nodes, 128 threads (4 waves). Wave w computes output cols
// [w*16, w*16+16) for all 4 M-tiles. `out` holds agg on entry; the final
// result overwrites it (each block only touches its own rows, reads first).
__launch_bounds__(128)
__global__ void gk_kan(const float* __restrict__ x,
                       float* __restrict__ out,   // agg in, h_layer+h_conv out
                       const __bf16* __restrict__ wbL, const __bf16* __restrict__ wsL,
                       const __bf16* __restrict__ wbC, const __bf16* __restrict__ wsC,
                       int nNodes) {
  __shared__ __align__(16) __bf16 sBase[TILE_M * N_IN];   // silu(in) :  8 KB
  __shared__ __align__(16) __bf16 sSpl [TILE_M * KSPL];   // bases    : 64 KB

  const int tid  = threadIdx.x;
  const int lane = tid & 31;
  const int nt   = tid >> 5;       // wave id == output N-tile
  const int nn   = lane & 15;
  const int hi   = lane >> 4;
  const int base = blockIdx.x * TILE_M;

  const float hstep = 0.4f;        // (GRID_MAX-GRID_MIN)/GRID_SIZE
  v8f acc[M_TILES];
  #pragma unroll
  for (int mt = 0; mt < M_TILES; ++mt) acc[mt] = v8f{};

  #pragma unroll
  for (int pass = 0; pass < 2; ++pass) {
    const float* in  = (pass == 0) ? x   : out;
    const __bf16* Wb = (pass == 0) ? wbL : wbC;
    const __bf16* Ws = (pass == 0) ? wsL : wsC;

    __syncthreads();   // previous pass done reading LDS
    // Fill LDS: 64 rows x 64 features; 32 (node,feature) pairs per thread.
    for (int p = tid; p < TILE_M * N_IN; p += 128) {
      int m = p >> 6, f = p & 63;
      int row = base + m;
      float v = (row < nNodes) ? in[(size_t)row * N_IN + f] : 0.0f;
      float s = v / (1.0f + __expf(-v));            // silu
      sBase[m * N_IN + f] = (__bf16)s;
      // Cox-de Boor cubic bases on uniform extended grid t_i = -1+(i-3)*h
      float b[11];
      #pragma unroll
      for (int i = 0; i < 11; ++i) {
        float t0 = -1.0f + (float)(i - 3) * hstep;
        b[i] = (v >= t0 && v < t0 + hstep) ? 1.0f : 0.0f;
      }
      #pragma unroll
      for (int j = 1; j <= 3; ++j) {
        float invj = 1.0f / ((float)j * hstep);
        #pragma unroll
        for (int i = 0; i < 11 - j; ++i) {
          float ti  = -1.0f + (float)(i - 3) * hstep;
          float tip = ti + (float)(j + 1) * hstep;
          b[i] = (v - ti) * invj * b[i] + (tip - v) * invj * b[i + 1];
        }
      }
      v8bf sp;
      #pragma unroll
      for (int c = 0; c < 8; ++c) sp[c] = (__bf16)b[c];
      *(v8bf*)(&sSpl[m * KSPL + f * 8]) = sp;       // ds_store_b128
    }
    __syncthreads();

    // base path: (64x64 @ 64x16) -> 2 K-steps x 4 M-tiles
    {
      const __bf16* brow = Wb + (size_t)(nt * 16 + nn) * N_IN;
      #pragma unroll
      for (int k0 = 0; k0 < N_IN; k0 += 32) {
        v16bf b = gk_load_b(brow, k0, hi);
        #pragma unroll
        for (int mt = 0; mt < M_TILES; ++mt) {
          v16bf a = gk_load_a(sBase + (size_t)(mt * 16 + nn) * N_IN, k0, hi);
          acc[mt] = __builtin_amdgcn_wmma_f32_16x16x32_bf16(
              false, a, false, b, (short)0, acc[mt], false, false);
        }
      }
    }
    // spline path: (64x512 @ 512x16) -> 16 K-steps x 4 M-tiles
    {
      const __bf16* brow = Ws + (size_t)(nt * 16 + nn) * KSPL;
      #pragma unroll 2
      for (int k0 = 0; k0 < KSPL; k0 += 32) {
        v16bf b = gk_load_b(brow, k0, hi);
        #pragma unroll
        for (int mt = 0; mt < M_TILES; ++mt) {
          v16bf a = gk_load_a(sSpl + (size_t)(mt * 16 + nn) * KSPL, k0, hi);
          acc[mt] = __builtin_amdgcn_wmma_f32_16x16x32_bf16(
              false, a, false, b, (short)0, acc[mt], false, false);
        }
      }
    }
  }

  // C/D layout: VGPR r holds M = r + hi*8, col N = nn (per ISA table).
  const int col = nt * 16 + nn;
  #pragma unroll
  for (int mt = 0; mt < M_TILES; ++mt) {
    #pragma unroll
    for (int r = 0; r < 8; ++r) {
      int row = base + mt * 16 + hi * 8 + r;
      if (row < nNodes) out[(size_t)row * N_OUT + col] = acc[mt][r];
    }
  }
}

// ------------------------------ launcher -----------------------------------

extern "C" void kernel_launch(void* const* d_in, const int* in_sizes, int n_in,
                              void* d_out, int out_size, void* d_ws, size_t ws_size,
                              hipStream_t stream) {
  const float* x   = (const float*)d_in[0];
  const float* bwL = (const float*)d_in[1];   // (64,64)
  const float* swL = (const float*)d_in[2];   // (64,64,8)
  const float* bwC = (const float*)d_in[3];
  const float* swC = (const float*)d_in[4];
  const int*   ei  = (const int*)d_in[5];     // (2,E) row-major: src row, dst row
  float* out = (float*)d_out;

  const int nNodes = in_sizes[0] / N_IN;
  const int nEdges = in_sizes[5] / 2;

  // Workspace: deg + dinv + bf16 weights (~1 MB total). agg lives in d_out.
  char* ws = (char*)d_ws;
  size_t off = 0;
  auto take = [&](size_t bytes) -> char* {
    char* p = ws + off;
    off += (bytes + 255) & ~(size_t)255;
    return p;
  };
  float*  deg  = (float*)take((size_t)nNodes * 4);
  float*  dinv = (float*)take((size_t)nNodes * 4);
  __bf16* wbL  = (__bf16*)take((size_t)N_OUT * N_IN * 2);
  __bf16* wsLb = (__bf16*)take((size_t)N_OUT * KSPL * 2);
  __bf16* wbC  = (__bf16*)take((size_t)N_OUT * N_IN * 2);
  __bf16* wsCb = (__bf16*)take((size_t)N_OUT * KSPL * 2);
  (void)ws_size; (void)n_in;

  // 1) zero agg (in d_out) and deg
  gk_zero_f32<<<2048, 256, 0, stream>>>(out, out_size);
  gk_zero_f32<<<(nNodes + 255) / 256, 256, 0, stream>>>(deg, nNodes);

  // 2) weights -> bf16
  gk_cvt_f32_bf16<<<(N_OUT * N_IN  + 255) / 256, 256, 0, stream>>>(bwL, wbL,  N_OUT * N_IN);
  gk_cvt_f32_bf16<<<(N_OUT * KSPL + 255) / 256, 256, 0, stream>>>(swL, wsLb, N_OUT * KSPL);
  gk_cvt_f32_bf16<<<(N_OUT * N_IN  + 255) / 256, 256, 0, stream>>>(bwC, wbC,  N_OUT * N_IN);
  gk_cvt_f32_bf16<<<(N_OUT * KSPL + 255) / 256, 256, 0, stream>>>(swC, wsCb, N_OUT * KSPL);

  // 3) degree -> dinv
  gk_degree<<<(nEdges + 255) / 256, 256, 0, stream>>>(ei, deg, nEdges);
  gk_dinv<<<(nNodes + 255) / 256, 256, 0, stream>>>(deg, dinv, nNodes);

  // 4) edge scatter-add into agg (d_out); 8 edges per 256-thread block
  gk_scatter<<<(nEdges + 7) / 8, 256, 0, stream>>>(x, ei, dinv, out, nEdges);

  // 5) fused KAN (x + layer W) + (agg + conv W), WMMA bf16, f32 accumulate
  gk_kan<<<(nNodes + TILE_M - 1) / TILE_M, 128, 0, stream>>>(
      x, out, wbL, wsLb, wbC, wsCb, nNodes);
}
